// CrossAttentionModel_70806830842263
// MI455X (gfx1250) — matrix operand order repaired and checked by reading, not compiled, not run on hardware
//
#include <hip/hip_runtime.h>

// ---------------------------------------------------------------------------
// CrossAttention model on gfx1250 (MI455X), wave32 + WMMA f16 path.
//   q = relu(STE_Q@Wq+bq); k = relu(STE_P@Wk+bk); v = relu(X@Wv+bv)
//   per (b, n, head): softmax(q kT / 8) v
//   out = relu(attn@W1+b1)@W2 + b2
// ---------------------------------------------------------------------------

typedef __attribute__((ext_vector_type(16))) _Float16 v16h;
typedef __attribute__((ext_vector_type(8)))  _Float16 v8h;
typedef __attribute__((ext_vector_type(8)))  float    v8f;
typedef __attribute__((ext_vector_type(4)))  float    v4f;
typedef __attribute__((vector_size(4 * sizeof(int)))) int v4i_vs;

#define DMODEL 512
#define TT     48          // Tq == Tp
#define NNODES 256
#define HEADS  8
#define DHEAD  64
#define NBATCH 8
#define ROWS   (NBATCH * TT * NNODES)   // 98304 rows for every GEMM

// --- CDNA5 async global->LDS copy (ASYNCcnt path), guarded by __has_builtin --
#if defined(__HIP_DEVICE_COMPILE__) && \
    __has_builtin(__builtin_amdgcn_global_load_async_to_lds_b128) && \
    __has_builtin(__builtin_amdgcn_s_wait_asynccnt)
#define USE_ASYNC_LDS 1
#endif

__device__ __forceinline__ void cp16_g2l(const void* g, void* l) {
#if USE_ASYNC_LDS
  __builtin_amdgcn_global_load_async_to_lds_b128(
      (__attribute__((address_space(1))) v4i_vs*)g,
      (__attribute__((address_space(3))) v4i_vs*)l, 0, 0);
#else
  *(v8h*)l = *(const v8h*)g;
#endif
}

__device__ __forceinline__ void async_wait() {
#if USE_ASYNC_LDS
  __builtin_amdgcn_s_wait_asynccnt(0);
#endif
}

__device__ __forceinline__ v16h cat16(v8h lo, v8h hi) {
  return __builtin_shufflevector(lo, hi, 0, 1, 2, 3, 4, 5, 6, 7,
                                         8, 9, 10, 11, 12, 13, 14, 15);
}

// ---------------------------------------------------------------------------
// Weight conversion f32 -> f16
// ---------------------------------------------------------------------------
__global__ void f32_to_f16_kernel(const float* __restrict__ s,
                                  _Float16* __restrict__ d, int n) {
  int i = blockIdx.x * blockDim.x + threadIdx.x;
  if (i < n) d[i] = (_Float16)s[i];
}

// ---------------------------------------------------------------------------
// GEMM: C[ROWS x 512] = act(A[ROWS x 512] @ W[512 x 512] + bias)
// Block: 256 threads (8 waves). Block tile 256(M) x 64(N), K-step 32.
// Wave w owns rows [w*32, w*32+32) x all 64 N cols: 8 WMMAs / K-step against
// 12 ds_load_b128 (2 A frags, 4 B frags). A staged to LDS as f16 (async DMA
// when A is already f16); W staged transposed so B fragments are contiguous.
// ---------------------------------------------------------------------------
template <bool A_F16, bool RELU, bool OUT_F16>
__global__ void __launch_bounds__(256)
gemm512_kernel(const void* __restrict__ Ain, const _Float16* __restrict__ W,
               const float* __restrict__ bias, void* __restrict__ Cout) {
  __shared__ __align__(16) _Float16 As[256][40];   // [m][k], pad 8
  __shared__ __align__(16) _Float16 BsT[64][40];   // [n][k], pad 8

  const int t    = threadIdx.x;
  const int wave = t >> 5;
  const int lane = t & 31;
  const int half = lane >> 4;
  const int l16  = lane & 15;
  const size_t m0 = (size_t)blockIdx.y * 256;
  const int    n0 = blockIdx.x * 64;

  v8f acc[2][4] = {};

  const int br = t >> 3;          // 0..31  : W tile row (k)
  const int bc = (t & 7) * 8;     // 0..56  : W tile col (n)

  for (int kk = 0; kk < DMODEL; kk += 32) {
    // ---- stage A tile (256 x 32): thread t stages row t ----
    if (A_F16) {
      const _Float16* Ap = (const _Float16*)Ain + (m0 + t) * DMODEL + kk;
#pragma unroll
      for (int i = 0; i < 4; ++i) cp16_g2l(Ap + 8 * i, &As[t][8 * i]);
    } else {
      const float* Ap = (const float*)Ain + (m0 + t) * DMODEL + kk;
#pragma unroll
      for (int u = 0; u < 2; ++u) {
        v4f f0 = *(const v4f*)(Ap + 16 * u + 0);
        v4f f1 = *(const v4f*)(Ap + 16 * u + 4);
        v4f f2 = *(const v4f*)(Ap + 16 * u + 8);
        v4f f3 = *(const v4f*)(Ap + 16 * u + 12);
        v8h h0, h1;
#pragma unroll
        for (int i = 0; i < 4; ++i) {
          h0[i]     = (_Float16)f0[i];
          h0[i + 4] = (_Float16)f1[i];
          h1[i]     = (_Float16)f2[i];
          h1[i + 4] = (_Float16)f3[i];
        }
        *(v8h*)&As[t][16 * u]     = h0;
        *(v8h*)&As[t][16 * u + 8] = h1;
      }
    }
    // ---- stage B tile (32 x 64), transposed into LDS ----
    {
      const _Float16* Wp = W + (size_t)(kk + br) * DMODEL + n0 + bc;
      v8h w = *(const v8h*)Wp;
#pragma unroll
      for (int i = 0; i < 8; ++i) BsT[bc + i][br] = w[i];
    }
    async_wait();
    __syncthreads();

    // ---- fragments + WMMA ----
    v16h af[2];
#pragma unroll
    for (int mi = 0; mi < 2; ++mi) {
      const int am = wave * 32 + mi * 16 + l16;
      // 16-bit A layout: lanes 0-15 -> K 0-7 & 16-23; lanes 16-31 -> 8-15 & 24-31
      af[mi] = cat16(*(const v8h*)&As[am][half * 8],
                     *(const v8h*)&As[am][16 + half * 8]);
    }
#pragma unroll
    for (int j = 0; j < 4; ++j) {
      // B layout: lanes 0-15 -> K 0-7 & 16-23; lanes 16-31 -> 8-15 & 24-31, col = lane&15
      const _Float16* bp = &BsT[j * 16 + l16][half * 8];
      v16h bf = cat16(*(const v8h*)bp, *(const v8h*)(bp + 16));
#pragma unroll
      for (int mi = 0; mi < 2; ++mi)
        acc[mi][j] = __builtin_amdgcn_wmma_f32_16x16x32_f16(
            false, af[mi], false, bf, (short)0, acc[mi][j], false, false);
    }
    __syncthreads();
  }

  // ---- epilogue: bias (+ReLU) + store ----
#pragma unroll
  for (int mi = 0; mi < 2; ++mi)
#pragma unroll
    for (int j = 0; j < 4; ++j) {
      const int col = n0 + j * 16 + l16;
      const float bv = bias[col];
#pragma unroll
      for (int r = 0; r < 8; ++r) {
        const size_t row = m0 + wave * 32 + mi * 16 + r + half * 8;
        float x = acc[mi][j][r] + bv;
        if (RELU) x = x > 0.0f ? x : 0.0f;
        if (OUT_F16) ((_Float16*)Cout)[row * DMODEL + col] = (_Float16)x;
        else         ((float*)Cout)[row * DMODEL + col]    = x;
      }
    }
}

// ---------------------------------------------------------------------------
// Fused attention for one (b, n, head): Q,K,V are [48 x 64] f16 tiles with
// row stride N*D in global memory. 96 threads = 3 waves; wave w owns 16 rows.
//   S = Q K^T (6 WMMA), f32 softmax with scale 1/8, O = P V (8 WMMA).
// ---------------------------------------------------------------------------
__global__ void __launch_bounds__(96)
attn48_kernel(const _Float16* __restrict__ Q, const _Float16* __restrict__ K,
              const _Float16* __restrict__ V, _Float16* __restrict__ O) {
  __shared__ __align__(16) _Float16 Qs[TT][72];       // [tq][d]
  __shared__ __align__(16) _Float16 Ks[TT][72];       // [tp][d]
  __shared__ __align__(16) _Float16 VsT[DHEAD][72];   // [d][tp], tp 48..63 = 0
  __shared__ __align__(16) float    Sf[TT][52];       // scores f32
  __shared__ __align__(16) _Float16 Ps[TT][72];       // probs, cols 48..63 = 0

  const int t    = threadIdx.x;
  const int wave = t >> 5;
  const int lane = t & 31;
  const int half = lane >> 4;
  const int l16  = lane & 15;
  const int h = blockIdx.x, n = blockIdx.y, b = blockIdx.z;

  const size_t RS   = (size_t)NNODES * DMODEL;  // stride between time steps
  const size_t base = ((size_t)b * TT * NNODES + n) * DMODEL + (size_t)h * DHEAD;

  // ---- stage Q, K (async DMA), V^T (manual transpose) ----
  {
    const int r = t >> 1;          // 0..47
    const int c = (t & 1) * 32;    // 0/32
    const _Float16* qp = Q + base + (size_t)r * RS + c;
    const _Float16* kp = K + base + (size_t)r * RS + c;
    const _Float16* vp = V + base + (size_t)r * RS + c;
#pragma unroll
    for (int i = 0; i < 4; ++i) {
      cp16_g2l(qp + 8 * i, &Qs[r][c + 8 * i]);
      cp16_g2l(kp + 8 * i, &Ks[r][c + 8 * i]);
    }
#pragma unroll
    for (int i = 0; i < 32; ++i) VsT[c + i][r] = vp[i];
    if (t < DHEAD) {
      for (int p = TT; p < 64; ++p) VsT[t][p] = (_Float16)0.0f;
    }
  }
  async_wait();
  __syncthreads();

  // ---- S = Q K^T for rows [wave*16, wave*16+16) ----
  {
    v8f sc[3] = {};
#pragma unroll
    for (int ks = 0; ks < DHEAD; ks += 32) {
      const int am = wave * 16 + l16;
      v16h af = cat16(*(const v8h*)&Qs[am][ks + half * 8],
                      *(const v8h*)&Qs[am][ks + 16 + half * 8]);
#pragma unroll
      for (int j = 0; j < 3; ++j) {
        const _Float16* bp = &Ks[j * 16 + l16][ks + half * 8];
        v16h bf = cat16(*(const v8h*)bp, *(const v8h*)(bp + 16));
        sc[j] = __builtin_amdgcn_wmma_f32_16x16x32_f16(
            false, af, false, bf, (short)0, sc[j], false, false);
      }
    }
#pragma unroll
    for (int j = 0; j < 3; ++j)
#pragma unroll
      for (int r = 0; r < 8; ++r)
        Sf[wave * 16 + r + half * 8][j * 16 + l16] = sc[j][r];
  }
  __syncthreads();

  // ---- softmax over tp (48 cols), scale folded in ----
  if (t < TT) {
    const float scale = 0.125f;  // 1/sqrt(64)
    float mx = -3.0e38f;
    for (int c = 0; c < TT; ++c) mx = fmaxf(mx, Sf[t][c]);
    float sum = 0.0f;
    for (int c = 0; c < TT; ++c) {
      float e = __expf((Sf[t][c] - mx) * scale);
      Sf[t][c] = e;
      sum += e;
    }
    const float inv = 1.0f / sum;
    for (int c = 0; c < TT; ++c) Ps[t][c] = (_Float16)(Sf[t][c] * inv);
    for (int c = TT; c < 64; ++c) Ps[t][c] = (_Float16)0.0f;  // pad K to 64
  }
  __syncthreads();

  // ---- O = P V for rows [wave*16, wave*16+16) ----
  {
    v8f oc[4] = {};
#pragma unroll
    for (int ks = 0; ks < 64; ks += 32) {
      const int am = wave * 16 + l16;
      v16h af = cat16(*(const v8h*)&Ps[am][ks + half * 8],
                      *(const v8h*)&Ps[am][ks + 16 + half * 8]);
#pragma unroll
      for (int j = 0; j < 4; ++j) {
        const _Float16* bp = &VsT[j * 16 + l16][ks + half * 8];
        v16h bf = cat16(*(const v8h*)bp, *(const v8h*)(bp + 16));
        oc[j] = __builtin_amdgcn_wmma_f32_16x16x32_f16(
            false, af, false, bf, (short)0, oc[j], false, false);
      }
    }
#pragma unroll
    for (int j = 0; j < 4; ++j)
#pragma unroll
      for (int r = 0; r < 8; ++r) {
        const int tq = wave * 16 + r + half * 8;
        O[base + (size_t)tq * RS + j * 16 + l16] = (_Float16)oc[j][r];
      }
  }
}

// ---------------------------------------------------------------------------
// Launch pipeline
// ---------------------------------------------------------------------------
extern "C" void kernel_launch(void* const* d_in, const int* in_sizes, int n_in,
                              void* d_out, int out_size, void* d_ws,
                              size_t ws_size, hipStream_t stream) {
  (void)in_sizes; (void)n_in; (void)out_size; (void)ws_size;

  const float* X     = (const float*)d_in[0];
  const float* STE_P = (const float*)d_in[1];
  const float* STE_Q = (const float*)d_in[2];
  const float* Wq = (const float*)d_in[3];  const float* bq = (const float*)d_in[4];
  const float* Wk = (const float*)d_in[5];  const float* bk = (const float*)d_in[6];
  const float* Wv = (const float*)d_in[7];  const float* bv = (const float*)d_in[8];
  const float* W1 = (const float*)d_in[9];  const float* b1 = (const float*)d_in[10];
  const float* W2 = (const float*)d_in[11]; const float* b2 = (const float*)d_in[12];

  char* ws = (char*)d_ws;
  const size_t WSZ = (size_t)DMODEL * DMODEL;  // 262144 weights per matrix
  _Float16* Wq16 = (_Float16*)(ws + 0 * WSZ * 2);
  _Float16* Wk16 = (_Float16*)(ws + 1 * WSZ * 2);
  _Float16* Wv16 = (_Float16*)(ws + 2 * WSZ * 2);
  _Float16* W116 = (_Float16*)(ws + 3 * WSZ * 2);
  _Float16* W216 = (_Float16*)(ws + 4 * WSZ * 2);

  const size_t SZ = (size_t)ROWS * DMODEL;     // 50,331,648 activations
  _Float16* q16 = (_Float16*)(ws + ((size_t)4 << 20));
  _Float16* k16 = q16 + SZ;
  _Float16* v16 = k16 + SZ;
  _Float16* a16 = v16 + SZ;
  _Float16* h16 = q16;   // FFN hidden reuses q buffer (dead after attention)

  // weights -> f16
  dim3 cg((unsigned)(WSZ / 256)), cb(256);
  f32_to_f16_kernel<<<cg, cb, 0, stream>>>(Wq, Wq16, (int)WSZ);
  f32_to_f16_kernel<<<cg, cb, 0, stream>>>(Wk, Wk16, (int)WSZ);
  f32_to_f16_kernel<<<cg, cb, 0, stream>>>(Wv, Wv16, (int)WSZ);
  f32_to_f16_kernel<<<cg, cb, 0, stream>>>(W1, W116, (int)WSZ);
  f32_to_f16_kernel<<<cg, cb, 0, stream>>>(W2, W216, (int)WSZ);

  dim3 gg(DMODEL / 64, ROWS / 256), gb(256);
  // QKV projections (f32 A, ReLU, f16 out)
  gemm512_kernel<false, true, true><<<gg, gb, 0, stream>>>(STE_Q, Wq16, bq, q16);
  gemm512_kernel<false, true, true><<<gg, gb, 0, stream>>>(STE_P, Wk16, bk, k16);
  gemm512_kernel<false, true, true><<<gg, gb, 0, stream>>>(X,     Wv16, bv, v16);

  // attention per (head, node, batch)
  dim3 ag(HEADS, NNODES, NBATCH), ab(96);
  attn48_kernel<<<ag, ab, 0, stream>>>(q16, k16, v16, a16);

  // FFN
  gemm512_kernel<true, true,  true ><<<gg, gb, 0, stream>>>(a16, W116, b1, h16);
  gemm512_kernel<true, false, false><<<gg, gb, 0, stream>>>(h16, W216, b2, d_out);
}
